// AttentionalFactorizationMachine_16466904613346
// MI455X (gfx1250) — compile-verified, matrix operand order checked by reading, not run on hardware
//
#include <hip/hip_runtime.h>
#include <hip/hip_bf16.h>

typedef __attribute__((ext_vector_type(16))) __bf16 v16bf;
typedef __attribute__((ext_vector_type(8)))  float  v8f;

#define F_FIELDS 33
#define DIMD 128
#define DIMA 128
#define NPAIR 528   /* = 33*32/2 = 33 tiles of 16 rows exactly */

// ---------------------------------------------------------------------------
// Pre-kernel: split fp32 W[a,d] (row-major A x D) into bf16 hi/lo, stored in
// WMMA B-fragment order so the main kernel loads 32 contiguous bytes per lane.
// B tile = [K=32, N=16]; element (kk,n): lane = (n&15) | (kk[3]<<4),
// slot j = (kk[4]<<3) | kk[2:0]  (mirrors ISA 7.12.2 16-bit operand layout).
// ---------------------------------------------------------------------------
__global__ __launch_bounds__(256) void afm_repack_w(const float* __restrict__ W,
                                                    __bf16* __restrict__ Wh,
                                                    __bf16* __restrict__ Wl) {
  int idx = blockIdx.x * 256 + threadIdx.x;
  if (idx >= DIMA * DIMD) return;
  int a = idx >> 7;          // output column (N dim of GEMM)
  int d = idx & 127;         // K dim
  float w = W[idx];
  __bf16 h = (__bf16)w;
  __bf16 l = (__bf16)(w - (float)h);
  int kc = d >> 5, kk = d & 31, nt = a >> 4;
  int lane = (a & 15) | (((kk >> 3) & 1) << 4);
  int j    = ((kk >> 4) << 3) | (kk & 7);
  int off  = ((kc * 8 + nt) * 32 + lane) * 16 + j;
  Wh[off] = h;
  Wl[off] = l;
}

// ---------------------------------------------------------------------------
// Main kernel: one workgroup per batch row, 8 waves (wave32).
// ---------------------------------------------------------------------------
__global__ __launch_bounds__(256) void afm_main(const float* __restrict__ gnn,
                                                const float* __restrict__ x,
                                                const float* __restrict__ bias,
                                                const __bf16* __restrict__ Wh,
                                                const __bf16* __restrict__ Wl,
                                                float* __restrict__ out) {
  __shared__ __attribute__((aligned(16))) float  xs[F_FIELDS * DIMD];
  __shared__ __attribute__((aligned(32))) __bf16 Whs[DIMA * DIMD];
  __shared__ __attribute__((aligned(32))) __bf16 Wls[DIMA * DIMD];
  __shared__ float gnn_s[DIMA];
  __shared__ float bias_s[DIMA];
  __shared__ float sc_s[NPAIR];
  __shared__ unsigned char rows_s[NPAIR];
  __shared__ unsigned char cols_s[NPAIR];
  __shared__ float red_s[256];

  const int tid = threadIdx.x;
  const int b   = blockIdx.x;

  // ---- stage inputs into LDS -------------------------------------------
  {
    const float4* xb = (const float4*)(x + (size_t)b * F_FIELDS * DIMD);
    float4* xs4 = (float4*)xs;
    for (int i = tid; i < (F_FIELDS * DIMD) / 4; i += 256) xs4[i] = xb[i];
    if (tid < DIMA)            gnn_s[tid]        = gnn[(size_t)b * DIMA + tid];
    else if (tid < 2 * DIMA)   bias_s[tid - 128] = bias[tid - 128];
    const uint4* wh4 = (const uint4*)Wh;
    const uint4* wl4 = (const uint4*)Wl;
    uint4* whs4 = (uint4*)Whs;
    uint4* wls4 = (uint4*)Wls;
    for (int i = tid; i < (DIMA * DIMD * 2) / 16; i += 256) {
      whs4[i] = wh4[i];
      wls4[i] = wl4[i];
    }
    if (tid < F_FIELDS) {                 // triu_indices(F, k=1) tables
      int i = tid;
      int base = i * (2 * F_FIELDS - i - 1) / 2;
      for (int j2 = i + 1; j2 < F_FIELDS; ++j2) {
        rows_s[base + j2 - i - 1] = (unsigned char)i;
        cols_s[base + j2 - i - 1] = (unsigned char)j2;
      }
    }
  }
  __syncthreads();

  // ---- GEMM + score phase (bf16x3 WMMA) --------------------------------
  const int wave = tid >> 5;
  const int lane = tid & 31;
  const int mrow = lane & 15;   // A-matrix row within M-tile (ISA layout)
  const int half = lane >> 4;   // which K-halves this lane holds

#pragma unroll 1
  for (int mt = wave; mt < F_FIELDS; mt += 8) {
    const int p = mt * 16 + mrow;
    const float* xr = &xs[(int)rows_s[p] * DIMD];
    const float* xc = &xs[(int)cols_s[p] * DIMD];

    // Build A fragments (hi/lo split) for 4 K-chunks of 32.
    v16bf Ah[4], Al[4];
#pragma unroll
    for (int kc = 0; kc < 4; ++kc) {
#pragma unroll
      for (int j = 0; j < 16; ++j) {
        int k   = ((j & 8) << 1) | (half << 3) | (j & 7);
        float v = xr[kc * 32 + k] * xc[kc * 32 + k];   // inner[p, d]
        __bf16 h = (__bf16)v;
        Ah[kc][j] = h;
        Al[kc][j] = (__bf16)(v - (float)h);
      }
    }

    float sp[8];
#pragma unroll
    for (int v = 0; v < 8; ++v) sp[v] = 0.f;

#pragma unroll 1
    for (int nt = 0; nt < 8; ++nt) {
      // Preload ALL B fragments for this N-tile first (one load clause, no
      // WAR-hazard NOPs between kc groups). Opaque offset defeats LICM so the
      // 64 B-regs live only inside this iteration (no spill).
      v16bf Bh[4], Bl[4];
#pragma unroll
      for (int kc = 0; kc < 4; ++kc) {
        int off = ((kc * 8 + nt) * 32 + lane) * 16;
        asm volatile("" : "+v"(off));      // defeat LICM; still ds_load_b128
        Bh[kc] = *(const v16bf*)&Whs[off];
        Bl[kc] = *(const v16bf*)&Wls[off];
      }
      // Two independent accumulator chains (kc 0,2 and kc 1,3).
      v8f acc0 = {}, acc1 = {};
#pragma unroll
      for (int kc = 0; kc < 4; kc += 2) {
        acc0 = __builtin_amdgcn_wmma_f32_16x16x32_bf16(false, Ah[kc],     false, Bh[kc],
                                                       (short)0, acc0, false, false);
        acc1 = __builtin_amdgcn_wmma_f32_16x16x32_bf16(false, Ah[kc + 1], false, Bh[kc + 1],
                                                       (short)0, acc1, false, false);
        acc0 = __builtin_amdgcn_wmma_f32_16x16x32_bf16(false, Ah[kc],     false, Bl[kc],
                                                       (short)0, acc0, false, false);
        acc1 = __builtin_amdgcn_wmma_f32_16x16x32_bf16(false, Ah[kc + 1], false, Bl[kc + 1],
                                                       (short)0, acc1, false, false);
        acc0 = __builtin_amdgcn_wmma_f32_16x16x32_bf16(false, Al[kc],     false, Bh[kc],
                                                       (short)0, acc0, false, false);
        acc1 = __builtin_amdgcn_wmma_f32_16x16x32_bf16(false, Al[kc + 1], false, Bh[kc + 1],
                                                       (short)0, acc1, false, false);
      }
      // fold bias + relu + gnn for this 16-column tile
      const int a   = nt * 16 + mrow;      // C/D layout: N = lane&15
      const float bi = bias_s[a];
      const float g  = gnn_s[a];
#pragma unroll
      for (int v = 0; v < 8; ++v) {
        float z = acc0[v] + acc1[v] + bi;
        z = z > 0.f ? z : 0.f;
        sp[v] += z * g;
      }
    }

    // reduce over the 16 N-lanes of each half; lane 0 / lane 16 own a row each
#pragma unroll
    for (int v = 0; v < 8; ++v) {
      float s = sp[v];
      s += __shfl_xor(s, 1, 32);
      s += __shfl_xor(s, 2, 32);
      s += __shfl_xor(s, 4, 32);
      s += __shfl_xor(s, 8, 32);
      if (mrow == 0) sc_s[mt * 16 + (half << 3) + v] = s;   // M = v + 8*half
    }
  }
  __syncthreads();

  // ---- softmax over 528 scores -----------------------------------------
  float lm = -3.4e38f;
  for (int p = tid; p < NPAIR; p += 256) lm = fmaxf(lm, sc_s[p]);
  red_s[tid] = lm;
  __syncthreads();
  for (int s = 128; s >= 1; s >>= 1) {
    if (tid < s) red_s[tid] = fmaxf(red_s[tid], red_s[tid + s]);
    __syncthreads();
  }
  const float mx = red_s[0];
  __syncthreads();
  float ls = 0.f;
  for (int p = tid; p < NPAIR; p += 256) {
    float e = __expf(sc_s[p] - mx);
    sc_s[p] = e;
    ls += e;
  }
  red_s[tid] = ls;
  __syncthreads();
  for (int s = 128; s >= 1; s >>= 1) {
    if (tid < s) red_s[tid] += red_s[tid + s];
    __syncthreads();
  }
  const float scale = 100.f / red_s[0];
  __syncthreads();

  // ---- attn_output: weighted sum of inner, split pairs over two halves --
  const int d   = tid & 127;
  const int grp = tid >> 7;          // 0 -> pairs [0,264), 1 -> [264,528)
  float acc = 0.f;
  for (int p = grp * (NPAIR / 2); p < (grp + 1) * (NPAIR / 2); ++p) {
    acc += sc_s[p] * xs[(int)rows_s[p] * DIMD + d] * xs[(int)cols_s[p] * DIMD + d];
  }
  red_s[tid] = acc;
  __syncthreads();

  float* ob = out + (size_t)b * (DIMA + DIMD);
  if (tid < DIMD) {
    ob[tid]        = gnn_s[tid];                                  // concat: gnn
    ob[DIMD + tid] = (red_s[tid] + red_s[tid + 128]) * scale;     // attn_out*100
  }
}

// ---------------------------------------------------------------------------
extern "C" void kernel_launch(void* const* d_in, const int* in_sizes, int n_in,
                              void* d_out, int out_size, void* d_ws, size_t ws_size,
                              hipStream_t stream) {
  (void)n_in; (void)out_size; (void)ws_size;
  const float* gnn = (const float*)d_in[0];   // [B, 128]
  const float* x   = (const float*)d_in[1];   // [B, 33, 128]
  const float* W   = (const float*)d_in[2];   // [128, 128]
  const float* bia = (const float*)d_in[3];   // [128]
  float* out = (float*)d_out;                 // [B, 256]
  const int B = in_sizes[0] / DIMA;

  __bf16* Wh = (__bf16*)d_ws;                 // 32 KB
  __bf16* Wl = Wh + DIMA * DIMD;              // 32 KB

  afm_repack_w<<<(DIMA * DIMD + 255) / 256, 256, 0, stream>>>(W, Wh, Wl);
  afm_main<<<B, 256, 0, stream>>>(gnn, x, bia, Wh, Wl, out);
}
